// ReconstructedANI_80032420593976
// MI455X (gfx1250) — matrix operand order repaired and checked by reading, not compiled, not run on hardware
//
#include <hip/hip_runtime.h>
#include <hip/hip_bf16.h>

typedef __attribute__((ext_vector_type(16))) _Float16 v16h;
typedef __attribute__((ext_vector_type(8)))  _Float16 v8h;
typedef __attribute__((ext_vector_type(8)))  float    v8f;

#define RCRF   5.2f
#define ETA_RF 16.0f
#define NSPEC  7
#define NSHF   16
#define RAD_F  (NSPEC * NSHF)   // 112 real radial features
#define K1PAD  128              // padded K for layer 1 (multiple of 32)
#define D1     256
#define D2     192
#define D3     160

#if defined(__has_builtin)
#if __has_builtin(__builtin_amdgcn_sched_barrier)
#define SCHED_BARRIER() __builtin_amdgcn_sched_barrier(0)
#endif
#endif
#ifndef SCHED_BARRIER
#define SCHED_BARRIER()
#endif

__constant__ float c_shf[NSHF] = {
    0.9f, 1.1687f, 1.4375f, 1.7062f, 1.975f, 2.2437f, 2.5125f, 2.7812f,
    3.05f, 3.3187f, 3.5875f, 3.8562f, 4.125f, 4.3937f, 4.6625f, 4.9312f};

// ---------------------------------------------------------------------------
// Deterministic counting sort of atoms by species (species of j is i-independent,
// so this is done once for the whole AEV pass). Sequential per species ->
// fixed permutation, no atomics, so float accumulation order is identical on
// every call.
// ---------------------------------------------------------------------------
__global__ void sort_species(const int* __restrict__ species,
                             int* __restrict__ perm, int* __restrict__ offs, int N) {
  __shared__ int cnt[8];
  const int s = threadIdx.x;  // blockDim.x == 8
  if (s < NSPEC) {
    int c = 0;
    for (int n = 0; n < N; ++n) c += (species[n] == s);
    cnt[s] = c;
  }
  __syncthreads();
  if (s == 0) {
    int acc = 0;
    for (int q = 0; q < NSPEC; ++q) { int c = cnt[q]; cnt[q] = acc; acc += c; }
    cnt[NSPEC] = acc;
    for (int q = 0; q < NSPEC + 1; ++q) offs[q] = cnt[q];
  }
  __syncthreads();
  if (s < NSPEC) {
    int w = cnt[s];
    for (int n = 0; n < N; ++n)
      if (species[n] == s) perm[w++] = n;
  }
}

// ---------------------------------------------------------------------------
// AEV radial features: one block (128 threads) per atom i, iterating j in
// species-sorted order. Phase A computes (d, fc) for a 128-wide sorted chunk
// into LDS (packed float2 -> single ds_load_b64 broadcast per neighbor in the
// hot loop); Phase B: feature thread (s = t>>4, r = t&15) scans ONLY its
// species' segment of the chunk (7x less scan work; v_exp count already
// minimal). Output f16 [N,128] (cols 112..127 zero) = WMMA A operand.
// ---------------------------------------------------------------------------
__global__ __launch_bounds__(128) void aev_kernel(
    const int* __restrict__ perm, const int* __restrict__ offs,
    const float* __restrict__ coords, _Float16* __restrict__ aev, int N) {
  __shared__ float2 sdf[128];
  __shared__ int    soff[8];

  const int i = blockIdx.x;
  const int t = threadIdx.x;
  if (t < 8) soff[t] = offs[t];
  __syncthreads();

  const float xi = coords[i * 3 + 0];
  const float yi = coords[i * 3 + 1];
  const float zi = coords[i * 3 + 2];

  const int   myS = t >> 4;
  const float shf = c_shf[t & 15];
  int lo_s = 0, hi_s = 0;
  if (t < RAD_F) { lo_s = soff[myS]; hi_s = soff[myS + 1]; }
  float acc = 0.f;

  for (int j0 = 0; j0 < N; j0 += 128) {
    __syncthreads();
    const int jj = perm[j0 + t];
    const float dx = xi - coords[jj * 3 + 0];
    const float dy = yi - coords[jj * 3 + 1];
    const float dz = zi - coords[jj * 3 + 2];
    const float sq = dx * dx + dy * dy + dz * dz;
    const float d  = sqrtf(sq);
    float fc = 0.f;
    if (sq > 0.f && d <= RCRF)
      fc = 0.5f * __cosf(3.14159265358979323846f * d / RCRF) + 0.5f;
    sdf[t] = make_float2(d, fc);
    __syncthreads();

    if (t < RAD_F) {
      const int lo = max(lo_s - j0, 0);
      const int hi = min(hi_s - j0, 128);
      for (int u = lo; u < hi; ++u) {
        const float2 df = sdf[u];
        if (df.y > 0.f) {
          const float dd = df.x - shf;
          acc += __expf(-ETA_RF * dd * dd) * df.y;
        }
      }
    }
  }
  aev[(long)i * K1PAD + t] = (t < RAD_F) ? (_Float16)acc : (_Float16)0.f;
}

// ---------------------------------------------------------------------------
// Weight conversion: f32 -> f16 (layer 1 truncates K 1008 -> 128: AEV cols
// 112..1007 are structurally zero, so those W1 columns are dead).
// ---------------------------------------------------------------------------
__global__ void conv_w1(const float* __restrict__ W1, _Float16* __restrict__ w1h) {
  const int row = blockIdx.x;   // s*256 + o
  const int k   = threadIdx.x;  // 0..127
  w1h[(long)row * K1PAD + k] =
      (k < RAD_F) ? (_Float16)W1[(long)row * 1008 + k] : (_Float16)0.f;
}

__global__ void conv_f32_to_f16(const float* __restrict__ src,
                                _Float16* __restrict__ dst, int n) {
  const int idx = blockIdx.x * 256 + threadIdx.x;
  if (idx < n) dst[idx] = (_Float16)src[idx];
}

// ---------------------------------------------------------------------------
// WMMA GEMM + bias + CELU.  Hout[s,m,o] = celu(sum_k A[s?,m,k]*W[s,o,k]+b[s,o])
// Block = 256 threads = 8 waves; each wave owns FOUR 16x16 MxO tiles sharing
// one B fragment per K-step. Double-buffered K pipeline enforced with
// __builtin_amdgcn_sched_barrier(0): the prefetch loads for step k+1 are
// pinned ABOVE the 4 WMMAs of step k, so the backend cannot sink them and
// re-coalesce the two fragment buffers (which previously produced
// s_wait_loadcnt 0x0 before every WMMA). All loads are one base address +
// compile-time immediate offsets (fit the signed 24-bit IOFFSET).
// ---------------------------------------------------------------------------
template <int K>
__global__ __launch_bounds__(256, 1) void gemm_celu_wmma(
    const _Float16* __restrict__ A, const _Float16* __restrict__ W,
    const float* __restrict__ bias, _Float16* __restrict__ Hout,
    int M, int O, long a_sstride) {
  constexpr int NSTEP = K / 32;
  const int lane = threadIdx.x & 31;
  const int wave = threadIdx.x >> 5;
  const int s    = blockIdx.z;
  const int m0   = (blockIdx.x * 8 + wave) * 64;   // 4 M-tiles per wave
  const int o0   = blockIdx.y * 16;
  const int hi   = lane >> 4;
  const int l15  = lane & 15;

  // Single base pointer per operand; every load below is base + constant.
  const _Float16* Wp = W + ((long)s * O + (o0 + l15)) * K + (hi ? 16 : 0);
  const _Float16* Ap = A + (long)s * a_sstride + (long)(m0 + l15) * K + (hi ? 16 : 0);

  v8f  c[4] = {};
  v16h bf[2];
  v16h af[2][4];

  auto loadB = [&](int step, int buf) {
    const v8h x0 = *reinterpret_cast<const v8h*>(Wp + step * 32);
    const v8h x1 = *reinterpret_cast<const v8h*>(Wp + step * 32 + 8);
#pragma unroll
    for (int q = 0; q < 8; ++q) { bf[buf][q] = x0[q]; bf[buf][8 + q] = x1[q]; }
  };
  auto loadA = [&](int step, int buf) {
#pragma unroll
    for (int r = 0; r < 4; ++r) {
      const v8h x0 = *reinterpret_cast<const v8h*>(Ap + r * 16 * K + step * 32);
      const v8h x1 = *reinterpret_cast<const v8h*>(Ap + r * 16 * K + step * 32 + 8);
#pragma unroll
      for (int q = 0; q < 8; ++q) { af[buf][r][q] = x0[q]; af[buf][r][8 + q] = x1[q]; }
    }
  };

  loadB(0, 0);
  loadA(0, 0);
  SCHED_BARRIER();
#pragma unroll
  for (int step = 0; step < NSTEP; ++step) {
    const int cur = step & 1;
    const int nxt = cur ^ 1;
    if (step + 1 < NSTEP) {  // prefetch next K-step; pinned above current WMMAs
      loadB(step + 1, nxt);
      loadA(step + 1, nxt);
    }
    SCHED_BARRIER();
#pragma unroll
    for (int r = 0; r < 4; ++r)
      c[r] = __builtin_amdgcn_wmma_f32_16x16x32_f16(
          /*neg_a=*/false, af[cur][r], /*neg_b=*/false, bf[cur],
          /*c_mod=*/(short)0, c[r], /*reuse_a=*/false, /*reuse_b=*/false);
    SCHED_BARRIER();
  }

  const int   col = o0 + l15;
  const float bv  = bias[s * O + col];
#pragma unroll
  for (int r = 0; r < 4; ++r) {
    const int rbase = m0 + r * 16 + (hi ? 8 : 0);
#pragma unroll
    for (int q = 0; q < 8; ++q) {
      const float x = c[r][q] + bv;
      const float y = (x > 0.f) ? x : 0.1f * (__expf(10.f * x) - 1.f);
      Hout[((long)s * M + (rbase + q)) * O + col] = (_Float16)y;
    }
  }
}

// ---------------------------------------------------------------------------
// Layer 4 (O=1) fused with the species gate: E[n] = h3[s_n,n,:].W4[s_n]+b4[s_n]
// h3 rows are 320B (16B-aligned) -> read via b128 vector loads.
// ---------------------------------------------------------------------------
__global__ void dot_gate(const _Float16* __restrict__ H3p,
                         const float* __restrict__ W4,
                         const float* __restrict__ b4,
                         const int* __restrict__ species,
                         float* __restrict__ E, int M) {
  const int n = blockIdx.x * 256 + threadIdx.x;
  if (n >= M) return;
  const int s = species[n];
  const _Float16* h = H3p + ((long)s * M + n) * D3;
  const float*    w = W4 + (long)s * D3;
  float acc = b4[s];
#pragma unroll
  for (int k8 = 0; k8 < D3; k8 += 8) {
    const v8h hv = *reinterpret_cast<const v8h*>(h + k8);
#pragma unroll
    for (int q = 0; q < 8; ++q) acc += (float)hv[q] * w[k8 + q];
  }
  E[n] = acc;
}

__global__ void reduce_sum(const float* __restrict__ E, float* __restrict__ out, int M) {
  __shared__ float sm[256];
  float acc = 0.f;
  for (int idx = threadIdx.x; idx < M; idx += 256) acc += E[idx];
  sm[threadIdx.x] = acc;
  __syncthreads();
  for (int off = 128; off > 0; off >>= 1) {
    if (threadIdx.x < off) sm[threadIdx.x] += sm[threadIdx.x + off];
    __syncthreads();
  }
  if (threadIdx.x == 0) out[0] = sm[0];
}

// ---------------------------------------------------------------------------
extern "C" void kernel_launch(void* const* d_in, const int* in_sizes, int n_in,
                              void* d_out, int out_size, void* d_ws, size_t ws_size,
                              hipStream_t stream) {
  const int*   species = (const int*)d_in[0];
  const float* coords  = (const float*)d_in[1];
  const float* W1      = (const float*)d_in[2];
  const float* b1      = (const float*)d_in[3];
  const float* W2      = (const float*)d_in[4];
  const float* b2      = (const float*)d_in[5];
  const float* W3      = (const float*)d_in[6];
  const float* b3      = (const float*)d_in[7];
  const float* W4      = (const float*)d_in[8];
  const float* b4      = (const float*)d_in[9];
  const int M = in_sizes[0];   // B*N = 2048 (multiple of 512)

  char* p = (char*)d_ws;
  auto carve = [&](size_t bytes) -> void* {
    void* r = (void*)p;
    p += (bytes + 255) & ~(size_t)255;
    return r;
  };
  _Float16* aev  = (_Float16*)carve((size_t)M * K1PAD * 2);
  _Float16* w1h  = (_Float16*)carve((size_t)NSPEC * D1 * K1PAD * 2);
  _Float16* w2h  = (_Float16*)carve((size_t)NSPEC * D2 * D1 * 2);
  _Float16* w3h  = (_Float16*)carve((size_t)NSPEC * D3 * D2 * 2);
  _Float16* h1   = (_Float16*)carve((size_t)NSPEC * M * D1 * 2);
  _Float16* h2   = (_Float16*)carve((size_t)NSPEC * M * D2 * 2);
  _Float16* h3   = (_Float16*)carve((size_t)NSPEC * M * D3 * 2);
  float*    E    = (float*)carve((size_t)M * 4);
  int*      perm = (int*)carve((size_t)M * 4);
  int*      offs = (int*)carve(8 * 4);

  // 0) one-time deterministic species sort (shared by all atoms' AEV loops)
  sort_species<<<1, 8, 0, stream>>>(species, perm, offs, M);

  // 1) AEV radial features (transcendental-bound; dominant cost)
  aev_kernel<<<M, 128, 0, stream>>>(perm, offs, coords, aev, M);

  // 2) f32 -> f16 weight staging
  conv_w1<<<NSPEC * D1, K1PAD, 0, stream>>>(W1, w1h);
  conv_f32_to_f16<<<(NSPEC * D2 * D1 + 255) / 256, 256, 0, stream>>>(W2, w2h, NSPEC * D2 * D1);
  conv_f32_to_f16<<<(NSPEC * D3 * D2 + 255) / 256, 256, 0, stream>>>(W3, w3h, NSPEC * D3 * D2);

  // 3) WMMA MLP (all 7 experts; 4 M-tiles per wave sharing one B frag,
  //    sched_barrier-enforced double-buffered K pipeline)
  dim3 g1(M / 512, D1 / 16, NSPEC);
  gemm_celu_wmma<K1PAD><<<g1, 256, 0, stream>>>(aev, w1h, b1, h1, M, D1, 0L);
  dim3 g2(M / 512, D2 / 16, NSPEC);
  gemm_celu_wmma<D1><<<g2, 256, 0, stream>>>(h1, w2h, b2, h2, M, D2, (long)M * D1);
  dim3 g3(M / 512, D3 / 16, NSPEC);
  gemm_celu_wmma<D2><<<g3, 256, 0, stream>>>(h2, w3h, b3, h3, M, D3, (long)M * D2);

  // 4) final dot + species gate + scalar reduction
  dot_gate<<<(M + 255) / 256, 256, 0, stream>>>(h3, W4, b4, species, E, M);
  reduce_sum<<<1, 256, 0, stream>>>(E, (float*)d_out, M);
}
